// QueuePAWSTransformer_70531952935512
// MI455X (gfx1250) — matrix operand order repaired and checked by reading, not compiled, not run on hardware
//
#include <hip/hip_runtime.h>

typedef __bf16 bf16_t;
typedef __attribute__((ext_vector_type(16))) __bf16 v16bf;
typedef __attribute__((ext_vector_type(8)))  __bf16 v8bf;
typedef __attribute__((ext_vector_type(8)))  float  v8f;

#define NQ 4096   // queries (anchor/positive rows)
#define KQ 8192   // queue length
#define DF 512    // feature dim
#define NL 128    // label dim
#define NH 8      // heads
#define BH 512    // half of labeled batch (1024/2)

// ---------- WMMA helpers (gfx1250 layouts) ----------
// 16-bit A fragment 16x32: lane L: row = L%16; element j -> k = (j&8)*2 + (L/16)*8 + (j&7)
//   => two contiguous 8-element (16B) loads at k0 and k0+16.
static __device__ __forceinline__ v16bf load_a_frag(const bf16_t* p) {
  v8bf lo = *(const v8bf*)p;
  v8bf hi = *(const v8bf*)(p + 16);
  v16bf r;
#pragma unroll
  for (int j = 0; j < 8; ++j) { r[j] = lo[j]; r[8 + j] = hi[j]; }
  return r;
}
// 16-bit B fragment 32x16: lane L: col = L%16; k order must match the A fragment's (table order or contiguous)
static __device__ __forceinline__ v8f wmma_bf16(v16bf a, v16bf b, v8f c) {
  return __builtin_amdgcn_wmma_f32_16x16x32_bf16(false, a, false, b, (short)0, c, false, false);
}

// ---------- 1) confidence mask + stable compaction (prefix scan) ----------
__global__ void mask_scan_kernel(const float* __restrict__ logits1,
                                 const float* __restrict__ logits2,
                                 int* __restrict__ sel, int* __restrict__ cnt) {
  const int b = blockIdx.x;
  const float* lg = b ? logits2 : logits1;
  int* selb = sel + b * NQ;
  __shared__ unsigned s[1024];
  const int tid = threadIdx.x;
  unsigned flags = 0, lsum = 0;
#pragma unroll
  for (int i = 0; i < 4; ++i) {
    const int row = tid * 4 + i;
    const float* p = lg + (size_t)row * NL;
    float m = p[0];
    for (int c = 1; c < NL; ++c) m = fmaxf(m, p[c]);
    if (m > 0.95f) { flags |= 1u << i; lsum++; }
  }
  s[tid] = lsum;
  __syncthreads();
  for (int off = 1; off < 1024; off <<= 1) {
    unsigned t = 0;
    if (tid >= off) t = s[tid - off];
    __syncthreads();
    if (tid >= off) s[tid] += t;
    __syncthreads();
  }
  unsigned excl = s[tid] - lsum;
#pragma unroll
  for (int i = 0; i < 4; ++i)
    if (flags & (1u << i)) selb[excl++] = tid * 4 + i;
  if (tid == 1023) cnt[b] = (int)s[1023];
}

// ---------- 2) fused FIFO gather: feature queue -> bf16 [2][KQ][DF] ----------
__global__ void gather_feat_kernel(const float* __restrict__ anchor,
                                   const float* __restrict__ positive,
                                   const float* __restrict__ lb_feat,
                                   const float* __restrict__ fq1,
                                   const float* __restrict__ fq2,
                                   const int* __restrict__ sel,
                                   const int* __restrict__ cnt,
                                   bf16_t* __restrict__ fqbf) {
  const int b = blockIdx.y, i = blockIdx.x;
  const int C = cnt[b];
  const float* src;
  if (i < C)            src = (b ? positive : anchor) + (size_t)sel[b * NQ + i] * DF;
  else if (i < C + BH)  src = lb_feat + (size_t)(b * BH + (i - C)) * DF;
  else                  src = (b ? fq2 : fq1) + (size_t)(i - C - BH) * DF;
  bf16_t* dst = fqbf + ((size_t)b * KQ + i) * DF;
  for (int c = threadIdx.x; c < DF; c += 128) dst[c] = (bf16_t)src[c];
}

// ---------- 3) fused FIFO gather, transposed: V^T bf16 [2][NL][KQ] ----------
__global__ void gather_vt_kernel(const float* __restrict__ logits1,
                                 const float* __restrict__ logits2,
                                 const float* __restrict__ lb_oh,
                                 const float* __restrict__ lq1,
                                 const float* __restrict__ lq2,
                                 const int* __restrict__ sel,
                                 const int* __restrict__ cnt,
                                 bf16_t* __restrict__ vt) {
  const int b = blockIdx.z, c = blockIdx.y;
  const int i = blockIdx.x * 256 + threadIdx.x;
  const int C = cnt[b];
  const float* src;
  if (i < C)            src = (b ? logits2 : logits1) + (size_t)sel[b * NQ + i] * NL;
  else if (i < C + BH)  src = lb_oh + (size_t)(b * BH + (i - C)) * NL;
  else                  src = (b ? lq2 : lq1) + (size_t)(i - C - BH) * NL;
  vt[((size_t)b * NL + c) * KQ + i] = (bf16_t)src[c];
}

// ---------- 4) f32 -> bf16 convert ----------
__global__ void cvt_kernel(const float* __restrict__ src, bf16_t* __restrict__ dst, int n) {
  int i = blockIdx.x * 256 + threadIdx.x;
  if (i < n) dst[i] = (bf16_t)src[i];
}

// ---------- 5) transpose+convert weight: Wt[n][k] = W[k][n], 512x512 ----------
__global__ void transpose_w_kernel(const float* __restrict__ w, bf16_t* __restrict__ wt) {
  int k = blockIdx.x * 16 + (threadIdx.x & 15);
  int n = blockIdx.y * 16 + (threadIdx.x >> 4);
  wt[(size_t)n * DF + k] = (bf16_t)w[(size_t)k * DF + n];
}

// ---------- 6) projection GEMM: Out[M,512] = A[M,512] @ W (via Wt), bf16 WMMA ----------
__global__ __launch_bounds__(32) void gemm_bf16_kernel(const bf16_t* __restrict__ A,
                                                       const bf16_t* __restrict__ Wt,
                                                       bf16_t* __restrict__ Out) {
  const int tm = blockIdx.x * 16, tn = blockIdx.y * 16;
  const int lane = threadIdx.x, l16 = lane & 15, lh = lane >> 4;
  v8f acc = {};
  const bf16_t* arow = A + (size_t)(tm + l16) * DF;
  const bf16_t* brow = Wt + (size_t)(tn + l16) * DF;
  for (int kc = 0; kc < DF; kc += 32) {
    v16bf a = *(const v16bf*)(arow + kc + lh * 16);
    v16bf bfrag = *(const v16bf*)(brow + kc + lh * 16);
    acc = wmma_bf16(a, bfrag, acc);
  }
  // D: row = r + 8*lh, col = l16
#pragma unroll
  for (int r = 0; r < 8; ++r)
    Out[(size_t)(tm + r + 8 * lh) * DF + tn + l16] = (bf16_t)acc[r];
}

// ---------- 7) flash attention (4 waves/block, 2 heads per wave) + fused out-proj ----------
// S^T = K Q^T so query = lane%16 in all fragments; softmax stats via shfl_xor(16);
// P^T D->B fragment conversion needs only a lane<->lane^16 exchange.
__global__ __launch_bounds__(128) void attn_kernel(const bf16_t* __restrict__ Qproj,  // [2][NQ][DF]
                                                   const bf16_t* __restrict__ Kproj,  // [2][KQ][DF]
                                                   const bf16_t* __restrict__ Vt,     // [2][NL][KQ]
                                                   const float* __restrict__ Wo,      // [NL][NL]
                                                   const float* __restrict__ bo,      // [NL]
                                                   float* __restrict__ out) {         // [2][NQ][NL]
  const int b = blockIdx.y;
  const int qbase = blockIdx.x * 16;
  const int w = threadIdx.x >> 5;            // wave id 0..3
  const int lane = threadIdx.x & 31, l16 = lane & 15, lh = (lane >> 4) & 1;
  const bf16_t* Qp = Qproj + (size_t)b * NQ * DF;
  const bf16_t* Kp = Kproj + (size_t)b * KQ * DF;
  const bf16_t* Vp = Vt + (size_t)b * NL * KQ;

  __shared__ float lacc[4][16][NL];          // per-wave partial head sums, 32 KB

  // zero own slice
  for (int i = lane; i < 16 * NL; i += 32) ((float*)lacc[w])[i] = 0.f;
  __syncthreads();

  const bf16_t* qrow = Qp + (size_t)(qbase + l16) * DF;

  for (int hh = 0; hh < 2; ++hh) {
    const int h = w + hh * 4;                // heads {w, w+4}
    // Q^T B-fragments for this head (two 32-wide k chunks of dh=64)
    v16bf bq0 = load_a_frag(qrow + h * 64 + lh * 8);
    v16bf bq1 = load_a_frag(qrow + h * 64 + 32 + lh * 8);

    v8f oacc[8];
#pragma unroll
    for (int c = 0; c < 8; ++c) oacc[c] = (v8f){};
    float m_run = -1e30f, l_run = 0.f;

    for (int kb = 0; kb < KQ; kb += 32) {
      // scores S^T: D row = key (vgpr+8*lh within 16-key subtile), col = query (l16)
      const bf16_t* k0 = Kp + (size_t)(kb + l16) * DF + h * 64 + lh * 8;
      const bf16_t* k1 = Kp + (size_t)(kb + 16 + l16) * DF + h * 64 + lh * 8;
      __builtin_prefetch(k0 + 32 * DF, 0, 1);   // next key chunk -> global_prefetch_b8
      __builtin_prefetch(k1 + 32 * DF, 0, 1);
      v8f s0 = {}, s1 = {};
      s0 = wmma_bf16(load_a_frag(k0), bq0, s0);
      s0 = wmma_bf16(load_a_frag(k0 + 32), bq1, s0);
      s1 = wmma_bf16(load_a_frag(k1), bq0, s1);
      s1 = wmma_bf16(load_a_frag(k1 + 32), bq1, s1);
#pragma unroll
      for (int r = 0; r < 8; ++r) { s0[r] *= 0.125f; s1[r] *= 0.125f; }  // dh^-0.5

      // per-query (per-lane-column) online softmax stats; partner = lane ^ 16
      float mloc = s0[0];
#pragma unroll
      for (int r = 0; r < 8; ++r) { mloc = fmaxf(mloc, s0[r]); mloc = fmaxf(mloc, s1[r]); }
      float mchunk = fmaxf(mloc, __shfl_xor(mloc, 16, 32));
      float mnew = fmaxf(m_run, mchunk);
      float fac = __expf(m_run - mnew);

      float p0[8], p1[8], lloc = 0.f;
#pragma unroll
      for (int r = 0; r < 8; ++r) {
        p0[r] = __expf(s0[r] - mnew);
        p1[r] = __expf(s1[r] - mnew);
        lloc += p0[r] + p1[r];
      }
      l_run = l_run * fac + (lloc + __shfl_xor(lloc, 16, 32));
      m_run = mnew;
#pragma unroll
      for (int c = 0; c < 8; ++c)
#pragma unroll
        for (int r = 0; r < 8; ++r) oacc[c][r] *= fac;

      // assemble P^T B-fragment: element j -> key = kb + lh*16 + j
      float x0[8], x1[8];
#pragma unroll
      for (int r = 0; r < 8; ++r) { x0[r] = __shfl_xor(p0[r], 16, 32); x1[r] = __shfl_xor(p1[r], 16, 32); }
      v16bf pb;
#pragma unroll
      for (int j = 0; j < 8; ++j) {
        float vlo = lh ? x1[j] : p0[j];       // j<8
        float vhi = lh ? p1[j] : x0[j];       // j>=8
        pb[j] = (bf16_t)vlo;
        pb[j + 8] = (bf16_t)vhi;
      }

      // O^T += V^T_tile @ P^T  (8 output-channel tiles of 16)
#pragma unroll
      for (int c = 0; c < 8; ++c) {
        const bf16_t* vrow = Vp + (size_t)(16 * c + l16) * KQ + kb + lh * 16;
        oacc[c] = wmma_bf16(*(const v16bf*)vrow, pb, oacc[c]);
      }
    }

    // head done: normalize by l, average (1/8), accumulate into own LDS slice
    float invl = 1.0f / (8.0f * l_run);
#pragma unroll
    for (int c = 0; c < 8; ++c)
#pragma unroll
      for (int r = 0; r < 8; ++r)
        lacc[w][l16][16 * c + r + 8 * lh] += oacc[c][r] * invl;   // disjoint per lane
  }
  __syncthreads();

  // reduce 4 wave slices in place (each element owned by one thread)
  for (int i = threadIdx.x; i < 16 * NL; i += 128) {
    int q = i >> 7, ch = i & (NL - 1);
    lacc[0][q][ch] = lacc[0][q][ch] + lacc[1][q][ch] + lacc[2][q][ch] + lacc[3][q][ch];
  }
  __syncthreads();

  // fused output projection: out[q, t] = sum_c lacc[0][q][c] * Wo[c][t] + bo[t]
  const int t = threadIdx.x;  // 0..127 output channel
  float acc[16];
#pragma unroll
  for (int q = 0; q < 16; ++q) acc[q] = bo[t];
  for (int c = 0; c < NL; ++c) {
    float wv = Wo[(size_t)c * NL + t];
#pragma unroll
    for (int q = 0; q < 16; ++q) acc[q] = fmaf(lacc[0][q][c], wv, acc[q]);
  }
#pragma unroll
  for (int q = 0; q < 16; ++q)
    out[((size_t)b * NQ + qbase + q) * NL + t] = acc[q];
}

extern "C" void kernel_launch(void* const* d_in, const int* in_sizes, int n_in,
                              void* d_out, int out_size, void* d_ws, size_t ws_size,
                              hipStream_t stream) {
  (void)in_sizes; (void)n_in; (void)out_size; (void)ws_size;
  const float* anchor   = (const float*)d_in[0];
  const float* positive = (const float*)d_in[1];
  const float* lb_feat  = (const float*)d_in[2];
  const float* lb_oh    = (const float*)d_in[3];
  const float* logits1  = (const float*)d_in[5];
  const float* logits2  = (const float*)d_in[6];
  const float* fq1      = (const float*)d_in[7];
  const float* fq2      = (const float*)d_in[8];
  const float* lq1      = (const float*)d_in[9];
  const float* lq2      = (const float*)d_in[10];
  const float* Wq       = (const float*)d_in[11];
  const float* Wk       = (const float*)d_in[12];
  const float* Wo       = (const float*)d_in[13];
  const float* bo       = (const float*)d_in[14];

  char* ws = (char*)d_ws;
  size_t off = 0;
  auto alloc = [&](size_t bytes) -> void* {
    void* p = ws + off;
    off = (off + bytes + 255) & ~(size_t)255;
    return p;
  };
  int*    sel   = (int*)   alloc(2ull * NQ * 4);
  int*    cnt   = (int*)   alloc(2ull * 4);
  bf16_t* fqbf  = (bf16_t*)alloc(2ull * KQ * DF * 2);   // gathered key features, bf16
  bf16_t* vt    = (bf16_t*)alloc(2ull * NL * KQ * 2);   // gathered V, transposed, bf16
  bf16_t* qxbf  = (bf16_t*)alloc(2ull * NQ * DF * 2);   // x in bf16
  bf16_t* wqt   = (bf16_t*)alloc((size_t)DF * DF * 2);  // Wq^T bf16
  bf16_t* wkt   = (bf16_t*)alloc((size_t)DF * DF * 2);  // Wk^T bf16
  bf16_t* qproj = (bf16_t*)alloc(2ull * NQ * DF * 2);
  bf16_t* kproj = (bf16_t*)alloc(2ull * KQ * DF * 2);

  // 1) masks + stable compaction
  mask_scan_kernel<<<2, 1024, 0, stream>>>(logits1, logits2, sel, cnt);
  // 2) gather queues
  gather_feat_kernel<<<dim3(KQ, 2), 128, 0, stream>>>(anchor, positive, lb_feat, fq1, fq2,
                                                      sel, cnt, fqbf);
  gather_vt_kernel<<<dim3(KQ / 256, NL, 2), 256, 0, stream>>>(logits1, logits2, lb_oh, lq1, lq2,
                                                              sel, cnt, vt);
  // 3) bf16 staging of queries and weights
  cvt_kernel<<<(NQ * DF + 255) / 256, 256, 0, stream>>>(anchor, qxbf, NQ * DF);
  cvt_kernel<<<(NQ * DF + 255) / 256, 256, 0, stream>>>(positive, qxbf + (size_t)NQ * DF, NQ * DF);
  transpose_w_kernel<<<dim3(DF / 16, DF / 16), 256, 0, stream>>>(Wq, wqt);
  transpose_w_kernel<<<dim3(DF / 16, DF / 16), 256, 0, stream>>>(Wk, wkt);
  // 4) WMMA projections
  for (int b = 0; b < 2; ++b) {
    gemm_bf16_kernel<<<dim3(NQ / 16, DF / 16), 32, 0, stream>>>(
        qxbf + (size_t)b * NQ * DF, wqt, qproj + (size_t)b * NQ * DF);
    gemm_bf16_kernel<<<dim3(KQ / 16, DF / 16), 32, 0, stream>>>(
        fqbf + (size_t)b * KQ * DF, wkt, kproj + (size_t)b * KQ * DF);
  }
  // 5) flash attention (WMMA scores + WMMA PV) with fused head-avg + output projection
  attn_kernel<<<dim3(NQ / 16, 2), 128, 0, stream>>>(qproj, kproj, vt, Wo, bo, (float*)d_out);
}
